// DirectAUModel_4827543241263
// MI455X (gfx1250) — compile-verified, matrix operand order checked by reading, not run on hardware
//
#include <hip/hip_runtime.h>

typedef _Float16 v16h __attribute__((ext_vector_type(16)));
typedef float    v8f  __attribute__((ext_vector_type(8)));

#define BATCH 8192
#define DIM   64
#define TILE  128
#define NTILE (BATCH / TILE)      // 64 tiles per dimension
#define LDSS  80                  // padded LDS row stride (halves): 160B = 32B-aligned rows

// Raw v_exp_f32 (no denormal-range guard). Arg range here is [-11.6, 0],
// far from the |x| >= 126 region where the libm guard matters.
__device__ __forceinline__ float fast_exp2(float x) {
#if __has_builtin(__builtin_amdgcn_exp2f)
    return __builtin_amdgcn_exp2f(x);
#else
    return exp2f(x);
#endif
}

// ---------------------------------------------------------------------------
// Kernel 1: gather rows, normalize, write f16 copies, per-row alignment term.
// One wave32 per batch row; lane handles 2 of the 64 dims.
// ---------------------------------------------------------------------------
__global__ void gather_normalize_kernel(const int* __restrict__ user_id,
                                        const int* __restrict__ pos_id,
                                        const float* __restrict__ user_table,
                                        const float* __restrict__ item_table,
                                        _Float16* __restrict__ Xu,
                                        _Float16* __restrict__ Xp,
                                        float* __restrict__ align_partial) {
    const int t    = threadIdx.x;
    const int wave = t >> 5;
    const int lane = t & 31;
    const int row  = blockIdx.x * 8 + wave;
    if (row >= BATCH) return;

    const int uid = user_id[row];
    const int pid = pos_id[row];
    const float2 u = *(const float2*)(user_table + (size_t)uid * DIM + 2 * lane);
    const float2 p = *(const float2*)(item_table + (size_t)pid * DIM + 2 * lane);

    float su = u.x * u.x + u.y * u.y;
    float sp = p.x * p.x + p.y * p.y;
    #pragma unroll
    for (int off = 16; off > 0; off >>= 1) {
        su += __shfl_xor(su, off, 32);
        sp += __shfl_xor(sp, off, 32);
    }
    const float ru = rsqrtf(su);
    const float rp = rsqrtf(sp);
    const float unx = u.x * ru, uny = u.y * ru;
    const float pnx = p.x * rp, pny = p.y * rp;

    union { _Float16 h[2]; unsigned int w; } cu, cp;
    cu.h[0] = (_Float16)unx; cu.h[1] = (_Float16)uny;
    cp.h[0] = (_Float16)pnx; cp.h[1] = (_Float16)pny;
    *(unsigned int*)(Xu + (size_t)row * DIM + 2 * lane) = cu.w;
    *(unsigned int*)(Xp + (size_t)row * DIM + 2 * lane) = cp.w;

    const float dx = unx - pnx, dy = uny - pny;
    float d = dx * dx + dy * dy;
    #pragma unroll
    for (int off = 16; off > 0; off >>= 1) d += __shfl_xor(d, off, 32);
    if (lane == 0) align_partial[row] = d;
}

// ---------------------------------------------------------------------------
// Kernel 2: fused Gram / exp / reduce with WMMA.
// Block = 256 threads (8 waves). Block (bi,bj,z) computes the 128x128 tile
// S = X[bi*128:..] * X[bj*128:..]^T for matrix z (0=user,1=pos), only bj>=bi,
// sums exp(4*(s-1)) over the tile (off-diagonal tiles weighted 2x), and
// writes one deterministic partial per block.
// ---------------------------------------------------------------------------
__global__ void gram_exp_kernel(const _Float16* __restrict__ Xu,
                                const _Float16* __restrict__ Xp,
                                float* __restrict__ unif_partial) {
    const int bi = blockIdx.x, bj = blockIdx.y, z = blockIdx.z;
    if (bj < bi) return;
    const _Float16* __restrict__ X = z ? Xp : Xu;

    __shared__ __align__(32) _Float16 As[TILE * LDSS];
    __shared__ __align__(32) _Float16 Bs[TILE * LDSS];
    __shared__ float wsum[8];

    const int t = threadIdx.x;
    // Each 128x64 f16 tile is a contiguous 16KB slab of X -> coalesced b128 copy.
    const uint4* __restrict__ srcA = (const uint4*)(X + (size_t)bi * TILE * DIM);
    const uint4* __restrict__ srcB = (const uint4*)(X + (size_t)bj * TILE * DIM);
    #pragma unroll
    for (int c = t; c < TILE * DIM / 8; c += 256) {
        const int r = c >> 3, o = c & 7;              // row, 8-half chunk in row
        *(uint4*)(As + r * LDSS + o * 8) = srcA[c];
        *(uint4*)(Bs + r * LDSS + o * 8) = srcB[c];
    }
    __syncthreads();

    const int wave = t >> 5, lane = t & 31;
    const int m0   = wave * 16;                       // 16-row strip per wave
    const int lrow = lane & 15;                       // M (or N) within 16
    const int koff = (lane >> 4) * 16;                // K half per lane group

    // A fragments for K=0..31 and K=32..63 (reused across all 8 column tiles)
    const v16h a0 = *(const v16h*)(As + (m0 + lrow) * LDSS + koff);
    const v16h a1 = *(const v16h*)(As + (m0 + lrow) * LDSS + koff + 32);

    const float C0 = 4.0f * 1.4426950408889634f;      // 4*log2(e)
    float esum = 0.0f;
    #pragma unroll
    for (int nt = 0; nt < 8; ++nt) {
        const v16h b0 = *(const v16h*)(Bs + (nt * 16 + lrow) * LDSS + koff);
        const v16h b1 = *(const v16h*)(Bs + (nt * 16 + lrow) * LDSS + koff + 32);
        v8f c = {};
        c = __builtin_amdgcn_wmma_f32_16x16x32_f16(false, a0, false, b0,
                                                   (short)0, c, false, false);
        c = __builtin_amdgcn_wmma_f32_16x16x32_f16(false, a1, false, b1,
                                                   (short)0, c, false, false);
        // exp(4(s-1)) = exp2(C0*s - C0): one v_fma_f32 + one v_exp_f32 each.
        #pragma unroll
        for (int i = 0; i < 8; ++i)
            esum += fast_exp2(fmaf(C0, c[i], -C0));
    }
    #pragma unroll
    for (int off = 16; off > 0; off >>= 1) esum += __shfl_xor(esum, off, 32);
    if (lane == 0) wsum[wave] = esum;
    __syncthreads();
    if (t == 0) {
        float s = 0.0f;
        #pragma unroll
        for (int i = 0; i < 8; ++i) s += wsum[i];
        const float wgt = (bi == bj) ? 1.0f : 2.0f;   // symmetry: count j>i twice
        unif_partial[z * (NTILE * NTILE) + bi * NTILE + bj] = s * wgt;
    }
}

// ---------------------------------------------------------------------------
// Kernel 3: deterministic finalize (fixed-order tree reductions, scalar out).
// ---------------------------------------------------------------------------
__global__ void finalize_kernel(const float* __restrict__ align_partial,
                                const float* __restrict__ unif_partial,
                                float* __restrict__ out) {
    __shared__ float red[256];
    const int t = threadIdx.x;

    float s = 0.0f;
    for (int i = t; i < BATCH; i += 256) s += align_partial[i];
    red[t] = s; __syncthreads();
    #pragma unroll
    for (int off = 128; off > 0; off >>= 1) {
        if (t < off) red[t] += red[t + off];
        __syncthreads();
    }
    const float align = red[0] / (float)BATCH;
    __syncthreads();

    const float n_pairs = 0.5f * (float)BATCH * (float)(BATCH - 1);
    float unif01 = 0.0f;
    for (int z = 0; z < 2; ++z) {
        s = 0.0f;
        for (int idx = t; idx < NTILE * NTILE; idx += 256) {
            const int i = idx >> 6, j = idx & (NTILE - 1);
            if (j >= i) s += unif_partial[z * NTILE * NTILE + idx];
        }
        red[t] = s; __syncthreads();
        #pragma unroll
        for (int off = 128; off > 0; off >>= 1) {
            if (t < off) red[t] += red[t + off];
            __syncthreads();
        }
        const float S = red[0];                       // full-matrix sum incl. diagonal
        __syncthreads();
        unif01 += logf(((S - (float)BATCH) * 0.5f) / n_pairs);
    }
    if (t == 0) out[0] = align + 1.0f /*GAMMA*/ * 0.5f * unif01;
}

// ---------------------------------------------------------------------------
extern "C" void kernel_launch(void* const* d_in, const int* in_sizes, int n_in,
                              void* d_out, int out_size, void* d_ws, size_t ws_size,
                              hipStream_t stream) {
    (void)in_sizes; (void)n_in; (void)out_size; (void)ws_size;
    const int*   user_id    = (const int*)d_in[0];
    const int*   pos_id     = (const int*)d_in[1];
    // d_in[2] = neg_id (unused by the reference computation)
    const float* user_table = (const float*)d_in[3];
    const float* item_table = (const float*)d_in[4];

    char* ws = (char*)d_ws;
    _Float16* Xu = (_Float16*)ws;                                  // 1 MB
    _Float16* Xp = (_Float16*)(ws + (size_t)BATCH * DIM * 2);      // 1 MB
    float* align_partial = (float*)(ws + (size_t)2 * BATCH * DIM * 2); // 32 KB
    float* unif_partial  = align_partial + BATCH;                  // 2*4096 f32

    gather_normalize_kernel<<<BATCH / 8, 256, 0, stream>>>(
        user_id, pos_id, user_table, item_table, Xu, Xp, align_partial);
    gram_exp_kernel<<<dim3(NTILE, NTILE, 2), 256, 0, stream>>>(
        Xu, Xp, unif_partial);
    finalize_kernel<<<1, 256, 0, stream>>>(align_partial, unif_partial,
                                           (float*)d_out);
}
